// VesselSAGE_42537356099859
// MI455X (gfx1250) — matrix operand order repaired and checked by reading, not compiled, not run on hardware
//
#include <hip/hip_runtime.h>

#define DD 128   // feature dim
#define NC 8     // classes
#define BN_EPS 1e-5f

typedef __bf16 bf16_t;
typedef bf16_t v16bf __attribute__((ext_vector_type(16)));
typedef float  v8f   __attribute__((ext_vector_type(8)));

// ---------------------------------------------------------------------------
// A-fragment loader: 16-bit A 16x32 layout (ISA 7.12.2):
//   lanes 0-15  hold row M=lane,    K = {k0+0..7,  k0+16..23}
//   lanes 16-31 hold row M=lane-16, K = {k0+8..15, k0+24..31}
// caller passes kk = k0 + (lane>=16 ? 8 : 0); values scaled by `mul` then
// converted f32 -> bf16.
// ---------------------------------------------------------------------------
__device__ __forceinline__ v16bf load_a_frag(const float* __restrict__ rowp,
                                             float mul, int kk) {
  float4 f0 = *(const float4*)(rowp + kk);
  float4 f1 = *(const float4*)(rowp + kk + 4);
  float4 f2 = *(const float4*)(rowp + kk + 16);
  float4 f3 = *(const float4*)(rowp + kk + 20);
  v16bf a;
  a[0]  = (bf16_t)(f0.x * mul); a[1]  = (bf16_t)(f0.y * mul);
  a[2]  = (bf16_t)(f0.z * mul); a[3]  = (bf16_t)(f0.w * mul);
  a[4]  = (bf16_t)(f1.x * mul); a[5]  = (bf16_t)(f1.y * mul);
  a[6]  = (bf16_t)(f1.z * mul); a[7]  = (bf16_t)(f1.w * mul);
  a[8]  = (bf16_t)(f2.x * mul); a[9]  = (bf16_t)(f2.y * mul);
  a[10] = (bf16_t)(f2.z * mul); a[11] = (bf16_t)(f2.w * mul);
  a[12] = (bf16_t)(f3.x * mul); a[13] = (bf16_t)(f3.y * mul);
  a[14] = (bf16_t)(f3.z * mul); a[15] = (bf16_t)(f3.w * mul);
  return a;
}

// ---------------------------------------------------------------------------
// degree: deg[dst] += 1  (f32 atomics, absorbed by 192MB L2)
// ---------------------------------------------------------------------------
__global__ void deg_kernel(const int* __restrict__ dst,
                           float* __restrict__ deg, int nE) {
  int e = blockIdx.x * blockDim.x + threadIdx.x;
  if (e < nE) atomicAdd(&deg[dst[e]], 1.0f);
}

__global__ void invdeg_kernel(const float* __restrict__ deg,
                              float* __restrict__ inv, int nN) {
  int i = blockIdx.x * blockDim.x + threadIdx.x;
  if (i < nN) {
    float d = deg[i];
    inv[i] = d > 0.0f ? 1.0f / d : 0.0f;
  }
}

// ---------------------------------------------------------------------------
// scatter-add aggregation: msg[dst,:] += x[src,:]
// 32 threads per edge, float4 per thread (b128 gather, 4 f32 atomics).
// ---------------------------------------------------------------------------
__global__ void scatter_kernel(const float* __restrict__ x,
                               const int* __restrict__ src,
                               const int* __restrict__ dst,
                               float* __restrict__ msg, int nE) {
  long long tid = (long long)blockIdx.x * blockDim.x + threadIdx.x;
  long long e = tid >> 5;
  if (e >= nE) return;
  int c = (int)(tid & 31);
  int s = src[e];
  int d = dst[e];
  float4 v = *(const float4*)(x + (size_t)s * DD + c * 4);
  float* mp = msg + (size_t)d * DD + c * 4;
  atomicAdd(mp + 0, v.x);
  atomicAdd(mp + 1, v.y);
  atomicAdd(mp + 2, v.z);
  atomicAdd(mp + 3, v.w);
}

// ---------------------------------------------------------------------------
// Fused SAGE layer:  xout = ReLU( BN( (msg*inv_deg)@Wl + x@Wr + bl ) )
// Single virtual GEMM [msg | x] (N x 256) @ [Wl ; Wr] (256 x 128).
// 256 threads = 8 wave32; each wave owns a 16-row x 128-col strip.
// Weights staged once per block into LDS, pre-swizzled into the
// V_WMMA_F32_16X16X32_BF16 B-fragment layout.
//
// Inner loop is software-pipelined:
//   - all 8 B fragments of K-step ks pulled into registers (clause of
//     ds_load_b128, single dscnt wait),
//   - A fragment for K-step ks+1 loaded from global + converted, hiding
//     HBM latency under the 8-WMMA burst,
//   - then 8 back-to-back v_wmma_f32_16x16x32_bf16.
// ---------------------------------------------------------------------------
__global__ __launch_bounds__(256) void sage_layer_kernel(
    const float* __restrict__ msg, const float* __restrict__ invdeg,
    const float* __restrict__ xin,
    const float* __restrict__ Wl, const float* __restrict__ Wr,
    const float* __restrict__ bl, const float* __restrict__ gamma,
    const float* __restrict__ beta, const float* __restrict__ mean,
    const float* __restrict__ var, float* __restrict__ xout, int nN) {
  // [kstep][coltile][lane][element] -> 8*8*32*16 bf16 = 64 KB LDS
  __shared__ __align__(32) bf16_t Bp[8][8][32][16];

  int tid = threadIdx.x;
  {
    // stage virtual weight row k = tid (0..127 -> Wl row, 128..255 -> Wr row)
    int k = tid;
    const float* wrow = (k < DD) ? (Wl + (size_t)k * DD)
                                 : (Wr + (size_t)(k - DD) * DD);
    int kstep = k >> 5;
    int kr = k & 31;
    int bhalf = (kr >> 3) & 1;  // which 8-block within each 16-block of K
    int e0 = (kr & 7) + ((kr >> 4) << 3);  // element index within fragment
    for (int col = 0; col < DD; ++col) {
      Bp[kstep][col >> 4][bhalf * 16 + (col & 15)][e0] = (bf16_t)wrow[col];
    }
  }
  __syncthreads();

  int lane = tid & 31;
  int wv = tid >> 5;
  int half = lane >> 4;
  int rowbase = blockIdx.x * 128 + wv * 16;
  int arow = rowbase + (lane & 15);
  if (arow >= nN) arow = nN - 1;  // clamp: clamped rows never stored

  float idg = invdeg[arow];
  const float* msgp = msg + (size_t)arow * DD;
  const float* xinp = xin + (size_t)arow * DD;
  __builtin_prefetch(xinp, 0, 0);

  v8f acc[8] = {};

  // prologue: A fragment for K-step 0 (msg half, scaled by inv_deg)
  v16bf a_cur = load_a_frag(msgp, idg, half * 8);

  for (int ks = 0; ks < 8; ++ks) {
    // pull all 8 B fragments of this K-step into registers first
    v16bf barr[8];
#pragma unroll
    for (int ct = 0; ct < 8; ++ct) {
      barr[ct] = *(const v16bf*)(&Bp[ks][ct][lane][0]);
    }
    // prefetch next K-step's A fragment (global b128 loads overlap WMMAs)
    v16bf a_next = a_cur;
    if (ks < 7) {
      int k0 = (ks + 1) * 32;
      const float* rowp = (k0 < DD) ? msgp : xinp;
      float mul = (k0 < DD) ? idg : 1.0f;
      a_next = load_a_frag(rowp, mul, (k0 & (DD - 1)) + half * 8);
    }
    // 8 back-to-back WMMAs on the current A fragment
#pragma unroll
    for (int ct = 0; ct < 8; ++ct) {
      acc[ct] = __builtin_amdgcn_wmma_f32_16x16x32_bf16(
          false, a_cur, false, barr[ct], (short)0, acc[ct], false, false);
    }
    a_cur = a_next;
  }

  // epilogue: C/D layout -> lane holds col = ct*16 + (lane&15),
  // VGPR r holds row = rowbase + r + 8*half.  Fold bl + BN into scale/shift.
#pragma unroll
  for (int ct = 0; ct < 8; ++ct) {
    int col = ct * 16 + (lane & 15);
    float s = gamma[col] * rsqrtf(var[col] + BN_EPS);
    float sh = (bl[col] - mean[col]) * s + beta[col];
#pragma unroll
    for (int r = 0; r < 8; ++r) {
      int orow = rowbase + r + half * 8;
      if (orow < nN) {
        float h = acc[ct][r] * s + sh;
        xout[(size_t)orow * DD + col] = h > 0.0f ? h : 0.0f;
      }
    }
  }
}

// ---------------------------------------------------------------------------
// Classifier: out = x @ Wc + bc  (Wc zero-padded 128x8 -> 128x16)
// ---------------------------------------------------------------------------
__global__ __launch_bounds__(256) void classifier_kernel(
    const float* __restrict__ xin, const float* __restrict__ Wc,
    const float* __restrict__ bc, float* __restrict__ out, int nN) {
  __shared__ __align__(32) bf16_t Bp[4][32][16];  // 4 KB
  int tid = threadIdx.x;
  if (tid < DD) {
    int k = tid;
    int kstep = k >> 5;
    int kr = k & 31;
    int bhalf = (kr >> 3) & 1;
    int e0 = (kr & 7) + ((kr >> 4) << 3);
    for (int col = 0; col < 16; ++col) {
      float w = (col < NC) ? Wc[(size_t)k * NC + col] : 0.0f;
      Bp[kstep][bhalf * 16 + (col & 15)][e0] = (bf16_t)w;
    }
  }
  __syncthreads();

  int lane = tid & 31;
  int wv = tid >> 5;
  int half = lane >> 4;
  int rowbase = blockIdx.x * 128 + wv * 16;
  int arow = rowbase + (lane & 15);
  if (arow >= nN) arow = nN - 1;

  const float* rowp = xin + (size_t)arow * DD;

  // preload all 4 B fragments, then burst the 4 WMMAs
  v16bf barr[4];
#pragma unroll
  for (int ks = 0; ks < 4; ++ks) {
    barr[ks] = *(const v16bf*)(&Bp[ks][lane][0]);
  }
  v8f acc = {};
#pragma unroll
  for (int ks = 0; ks < 4; ++ks) {
    v16bf a = load_a_frag(rowp, 1.0f, ks * 32 + half * 8);
    acc = __builtin_amdgcn_wmma_f32_16x16x32_bf16(
        false, a, false, barr[ks], (short)0, acc, false, false);
  }

  int col = lane & 15;
  if (col < NC) {
    float bias = bc[col];
#pragma unroll
    for (int r = 0; r < 8; ++r) {
      int orow = rowbase + r + half * 8;
      if (orow < nN) out[(size_t)orow * NC + col] = acc[r] + bias;
    }
  }
}

// ---------------------------------------------------------------------------
// Launch sequence (graph-capture safe: only async ops on `stream`).
// Workspace layout (floats):
//   deg[pad nN] | inv[pad nN] | msg[nN*128] | xa[nN*128] | xb[nN*128]
// ---------------------------------------------------------------------------
extern "C" void kernel_launch(void* const* d_in, const int* in_sizes, int n_in,
                              void* d_out, int out_size, void* d_ws,
                              size_t ws_size, hipStream_t stream) {
  const float* x0      = (const float*)d_in[0];
  const int*   eidx    = (const int*)d_in[1];
  const float* Wl      = (const float*)d_in[2];
  const float* bl      = (const float*)d_in[3];
  const float* Wr      = (const float*)d_in[4];
  const float* gamma   = (const float*)d_in[5];
  const float* beta    = (const float*)d_in[6];
  const float* bn_mean = (const float*)d_in[7];
  const float* bn_var  = (const float*)d_in[8];
  const float* Wc      = (const float*)d_in[9];
  const float* bc      = (const float*)d_in[10];

  int nN = in_sizes[0] / DD;
  int nE = in_sizes[1] / 2;
  const int* src = eidx;
  const int* dst = eidx + nE;

  size_t nNpad = ((size_t)nN + 1023) & ~(size_t)1023;
  float* deg = (float*)d_ws;
  float* inv = deg + nNpad;
  float* msg = inv + nNpad;
  float* xa  = msg + (size_t)nN * DD;
  float* xb  = xa + (size_t)nN * DD;

  const int tpb = 256;

  hipMemsetAsync(deg, 0, nNpad * sizeof(float), stream);
  deg_kernel<<<(nE + tpb - 1) / tpb, tpb, 0, stream>>>(dst, deg, nE);
  invdeg_kernel<<<(nN + tpb - 1) / tpb, tpb, 0, stream>>>(deg, inv, nN);

  const float* xcur = x0;
  float* bufs[2] = {xa, xb};
  for (int layer = 0; layer < 4; ++layer) {
    hipMemsetAsync(msg, 0, (size_t)nN * DD * sizeof(float), stream);
    long long sthreads = (long long)nE * 32;
    scatter_kernel<<<(int)((sthreads + tpb - 1) / tpb), tpb, 0, stream>>>(
        xcur, src, dst, msg, nE);
    float* xnext = bufs[layer & 1];
    sage_layer_kernel<<<(nN + 127) / 128, 256, 0, stream>>>(
        msg, inv, xcur,
        Wl + (size_t)layer * DD * DD, Wr + (size_t)layer * DD * DD,
        bl + (size_t)layer * DD, gamma + (size_t)layer * DD,
        beta + (size_t)layer * DD, bn_mean + (size_t)layer * DD,
        bn_var + (size_t)layer * DD, xnext, nN);
    xcur = xnext;
  }

  classifier_kernel<<<(nN + 127) / 128, 256, 0, stream>>>(
      xcur, Wc, bc, (float*)d_out, nN);
}
